// MultiHeadedAttention_24764781429542
// MI455X (gfx1250) — matrix operand order repaired and checked by reading, not compiled, not run on hardware
//
#include <hip/hip_runtime.h>

typedef __attribute__((ext_vector_type(16))) __bf16 v16bf;
typedef __attribute__((ext_vector_type(8)))  __bf16 v8bf;
typedef __attribute__((ext_vector_type(4)))  __bf16 v4bf;
typedef __attribute__((ext_vector_type(8)))  float  v8f;

constexpr int Tt  = 2048;
constexpr int Bb  = 2;
constexpr int Cc  = 1024;
constexpr int Hh  = 16;
constexpr int DKk = 64;

__device__ inline __bf16 f2bf(float f) { return (__bf16)f; }

__device__ inline v8f zero8() {
  v8f z;
#pragma unroll
  for (int i = 0; i < 8; ++i) z[i] = 0.0f;
  return z;
}

// D = A(16x32 bf16) * B(32x16 bf16) + C(16x16 f32)
__device__ inline v8f wmma_bf16(v16bf a, v16bf b, v8f c) {
  return __builtin_amdgcn_wmma_f32_16x16x32_bf16(
      false, a, false, b, (short)0, c, false, false);
}

__device__ inline v16bf combine16(v8bf lo, v8bf hi) {
  v16bf r;
#pragma unroll
  for (int i = 0; i < 8; ++i) { r[i] = lo[i]; r[8 + i] = hi[i]; }
  return r;
}

// A fragment (16x32, row-in-lane): lane's 16 elements are 16 contiguous
// K values: k = k0+16*half+0..15 (matches loadB16).
__device__ inline v16bf loadA16(const __bf16* __restrict__ row, int k0, int half) {
  v8bf lo = *reinterpret_cast<const v8bf*>(row + k0 + 16 * half);
  v8bf hi = *reinterpret_cast<const v8bf*>(row + k0 + 16 * half + 8);
  return combine16(lo, hi);
}

// B fragment (32x16, column-per-lane): 16 contiguous K values per lane
// starting at k0 + 16*half of the column's source row.
__device__ inline v16bf loadB16(const __bf16* __restrict__ row, int k0, int half) {
  const __bf16* p = row + k0 + 16 * half;
  v8bf lo = *reinterpret_cast<const v8bf*>(p);
  v8bf hi = *reinterpret_cast<const v8bf*>(p + 8);
  return combine16(lo, hi);
}

__device__ inline float rmax16(float v) {
#pragma unroll
  for (int off = 1; off < 16; off <<= 1)
    v = fmaxf(v, __shfl_xor(v, off, 32));
  return v;
}

__device__ inline float rsum16(float v) {
#pragma unroll
  for (int off = 1; off < 16; off <<= 1)
    v += __shfl_xor(v, off, 32);
  return v;
}

// ---------------------------------------------------------------------------
// fp32 -> bf16 row converter. permute=1: source rows (t*B+b) -> dest rows
// (b*T+t) (folds the (T,B,C)->(B,T,C) transpose into the conversion).
// ---------------------------------------------------------------------------
__global__ __launch_bounds__(256) void cvt_rows(
    const float* __restrict__ src, __bf16* __restrict__ dst,
    int nrows, int permute) {
  const int perRow = Cc / 4;
  const int idx = blockIdx.x * blockDim.x + threadIdx.x;
  const int row = idx / perRow;
  const int c4  = idx - row * perRow;
  if (row >= nrows) return;
  const int drow = permute ? ((row & (Bb - 1)) * Tt + (row >> 1)) : row;
  float4 f = reinterpret_cast<const float4*>(src)[idx];
  v4bf o;
  o[0] = f2bf(f.x); o[1] = f2bf(f.y); o[2] = f2bf(f.z); o[3] = f2bf(f.w);
  *reinterpret_cast<v4bf*>(dst + (size_t)drow * Cc + c4 * 4) = o;
}

// ---------------------------------------------------------------------------
// GEMM: Y[m,n] = sum_k A[m,k] * W[n,k] + bias[n]; A,W bf16 row-major [.,C].
// outMode 0: bf16 at [b,h,t,d]   outMode 1: bf16 at [b,h,d,t] (V^T)
// outMode 2: fp32 at (T,B,C)
// 4 waves/block, 32x64 tile/wave (2 A-tiles x 4 B-tiles = 8 WMMA / 12 loads
// per k-step); k-loop unrolled x2 so prefetch buffers alternate without
// register copies.
// ---------------------------------------------------------------------------
__global__ __launch_bounds__(128) void gemm_bf16(
    const __bf16* __restrict__ A, const __bf16* __restrict__ W,
    const float* __restrict__ bias, float scale, int outMode,
    __bf16* __restrict__ outB, float* __restrict__ outF) {
  const int lane = threadIdx.x & 31;
  const int wave = threadIdx.x >> 5;
  const int m    = lane & 15;
  const int half = lane >> 4;
  const int mbase = blockIdx.x * 128 + wave * 32;
  const int nbase = blockIdx.y * 64;

  const __bf16* arow0 = A + (size_t)(mbase + m) * Cc;
  const __bf16* arow1 = A + (size_t)(mbase + 16 + m) * Cc;
  const __bf16* wrow[4];
#pragma unroll
  for (int nt = 0; nt < 4; ++nt)
    wrow[nt] = W + (size_t)(nbase + nt * 16 + m) * Cc;

  v8f acc[2][4];
#pragma unroll
  for (int at = 0; at < 2; ++at)
#pragma unroll
    for (int nt = 0; nt < 4; ++nt) acc[at][nt] = zero8();

  // Buffer 0: k = 0
  v16bf a0[2] = { loadA16(arow0, 0, half), loadA16(arow1, 0, half) };
  v16bf b0[4];
#pragma unroll
  for (int nt = 0; nt < 4; ++nt) b0[nt] = loadB16(wrow[nt], 0, half);

  // Unroll-by-2 pipelined k loop (Cc/32 = 32 steps, even)
  for (int k0 = 0; k0 < Cc - 64; k0 += 64) {
    v16bf a1[2] = { loadA16(arow0, k0 + 32, half), loadA16(arow1, k0 + 32, half) };
    v16bf b1[4];
#pragma unroll
    for (int nt = 0; nt < 4; ++nt) b1[nt] = loadB16(wrow[nt], k0 + 32, half);
#pragma unroll
    for (int at = 0; at < 2; ++at)
#pragma unroll
      for (int nt = 0; nt < 4; ++nt)
        acc[at][nt] = wmma_bf16(a0[at], b0[nt], acc[at][nt]);

    a0[0] = loadA16(arow0, k0 + 64, half);
    a0[1] = loadA16(arow1, k0 + 64, half);
#pragma unroll
    for (int nt = 0; nt < 4; ++nt) b0[nt] = loadB16(wrow[nt], k0 + 64, half);
#pragma unroll
    for (int at = 0; at < 2; ++at)
#pragma unroll
      for (int nt = 0; nt < 4; ++nt)
        acc[at][nt] = wmma_bf16(a1[at], b1[nt], acc[at][nt]);
  }
  {  // tail: k = Cc-64 (in a0/b0) and Cc-32
    v16bf a1[2] = { loadA16(arow0, Cc - 32, half), loadA16(arow1, Cc - 32, half) };
    v16bf b1[4];
#pragma unroll
    for (int nt = 0; nt < 4; ++nt) b1[nt] = loadB16(wrow[nt], Cc - 32, half);
#pragma unroll
    for (int at = 0; at < 2; ++at)
#pragma unroll
      for (int nt = 0; nt < 4; ++nt)
        acc[at][nt] = wmma_bf16(a0[at], b0[nt], acc[at][nt]);
#pragma unroll
    for (int at = 0; at < 2; ++at)
#pragma unroll
      for (int nt = 0; nt < 4; ++nt)
        acc[at][nt] = wmma_bf16(a1[at], b1[nt], acc[at][nt]);
  }

  // Epilogue: C/D layout row = i + 8*half, col = lane&15 per 16x16 tile
#pragma unroll
  for (int at = 0; at < 2; ++at) {
#pragma unroll
    for (int nt = 0; nt < 4; ++nt) {
      const int n  = nbase + nt * 16 + m;
      const float bz = bias[n];
#pragma unroll
      for (int i = 0; i < 8; ++i) {
        const int row = mbase + at * 16 + i + 8 * half;  // = b*T + t
        const int t = row & (Tt - 1);
        const int b = row >> 11;
        const float val = (acc[at][nt][i] + bz) * scale;
        if (outMode == 0) {
          const int h = n >> 6, d = n & 63;
          outB[(((size_t)b * Hh + h) * Tt + t) * DKk + d] = f2bf(val);
        } else if (outMode == 1) {
          const int h = n >> 6, d = n & 63;
          outB[(((size_t)b * Hh + h) * DKk + d) * Tt + t] = f2bf(val);
        } else {
          outF[((size_t)t * Bb + b) * Cc + n] = val;
        }
      }
    }
  }
}

// ---------------------------------------------------------------------------
// One 32-key chunk of flash attention. Consumes kb0..kb3 (K tiles for c0),
// prefetches the next chunk's K tiles into nk0..nk3 between the S-WMMAs and
// the softmax VALU work.
// ---------------------------------------------------------------------------
__device__ inline void attn_chunk(
    int c0, int m, int half,
    v16bf qa0, v16bf qa1,
    v16bf kb0, v16bf kb1, v16bf kb2, v16bf kb3,
    int nc0, v16bf& nk0, v16bf& nk1, v16bf& nk2, v16bf& nk3,
    const __bf16* __restrict__ Kp, const __bf16* __restrict__ Vp,
    const unsigned char* __restrict__ mp,
    __bf16 (&Plds)[16][32],
    v8f& O0, v8f& O1, v8f& O2, v8f& O3,
    float* mrow, float* lrow) {
  // S = Q * K^T for this 16x32 chunk
  v8f S0 = zero8(), S1 = zero8();
  S0 = wmma_bf16(qa0, kb0, S0);
  S0 = wmma_bf16(qa1, kb1, S0);
  S1 = wmma_bf16(qa0, kb2, S1);
  S1 = wmma_bf16(qa1, kb3, S1);

  // V tiles for this chunk (consumed after softmax)
  v16bf vb0 = loadB16(Vp + (size_t)(0 * 16 + m) * Tt, c0, half);
  v16bf vb1 = loadB16(Vp + (size_t)(1 * 16 + m) * Tt, c0, half);
  v16bf vb2 = loadB16(Vp + (size_t)(2 * 16 + m) * Tt, c0, half);
  v16bf vb3 = loadB16(Vp + (size_t)(3 * 16 + m) * Tt, c0, half);

  // Prefetch next chunk's K tiles
  if (nc0 < Tt) {
    nk0 = loadB16(Kp + (size_t)(nc0 + m)      * DKk, 0,  half);
    nk1 = loadB16(Kp + (size_t)(nc0 + m)      * DKk, 32, half);
    nk2 = loadB16(Kp + (size_t)(nc0 + 16 + m) * DKk, 0,  half);
    nk3 = loadB16(Kp + (size_t)(nc0 + 16 + m) * DKk, 32, half);
  }

  const bool msk0 = mp[c0 + m] != 0;
  const bool msk1 = mp[c0 + 16 + m] != 0;

  // Online softmax per row r (+8*half); rows live in 16-lane groups
#pragma unroll
  for (int r = 0; r < 8; ++r) {
    const float s0 = msk0 ? -1e30f : S0[r];
    const float s1 = msk1 ? -1e30f : S1[r];
    const float cm = rmax16(fmaxf(s0, s1));
    const float mn = fmaxf(mrow[r], cm);
    const float sc = __expf(mrow[r] - mn);
    mrow[r] = mn;
    const float p0 = msk0 ? 0.0f : __expf(s0 - mn);
    const float p1 = msk1 ? 0.0f : __expf(s1 - mn);
    lrow[r] = lrow[r] * sc + rsum16(p0 + p1);
    O0[r] *= sc; O1[r] *= sc; O2[r] *= sc; O3[r] *= sc;
    Plds[r + 8 * half][m]      = f2bf(p0);
    Plds[r + 8 * half][16 + m] = f2bf(p1);
  }
  __syncthreads();

  // Re-read P in A layout: 16 contiguous keys 16*half+0..15 from LDS
  v8bf lo = *reinterpret_cast<const v8bf*>(&Plds[m][16 * half]);
  v8bf hi = *reinterpret_cast<const v8bf*>(&Plds[m][16 * half + 8]);
  v16bf pa = combine16(lo, hi);
  __syncthreads();

  // O += P * V
  O0 = wmma_bf16(pa, vb0, O0);
  O1 = wmma_bf16(pa, vb1, O1);
  O2 = wmma_bf16(pa, vb2, O2);
  O3 = wmma_bf16(pa, vb3, O3);
}

// ---------------------------------------------------------------------------
// Flash attention: one wave per 16-query tile per (b,h); online softmax,
// keys in chunks of 32; chunk loop unrolled x2 so the K double-buffer
// alternates without register copies.
// ---------------------------------------------------------------------------
__global__ __launch_bounds__(32) void flash_attn(
    const __bf16* __restrict__ Qw, const __bf16* __restrict__ Kw,
    const __bf16* __restrict__ Vw, const unsigned char* __restrict__ kpm,
    __bf16* __restrict__ ctx) {
  const int lane = threadIdx.x;
  const int m    = lane & 15;
  const int half = lane >> 4;
  const int bh   = blockIdx.y;
  const int b    = bh / Hh;
  const int h    = bh % Hh;
  const int q0   = blockIdx.x * 16;

  const __bf16* Qp = Qw + ((size_t)bh * Tt + q0) * DKk;  // [T][DK]
  const __bf16* Kp = Kw + (size_t)bh * Tt * DKk;         // [T][DK]
  const __bf16* Vp = Vw + (size_t)bh * DKk * Tt;         // [DK][T]
  const unsigned char* mp = kpm + (size_t)b * Tt;

  // Q fragments for d = 0..31 / 32..63, held across the whole key sweep
  const __bf16* qrow = Qp + (size_t)m * DKk;
  v16bf qa0 = loadA16(qrow, 0, half);
  v16bf qa1 = loadA16(qrow, 32, half);

  v8f O0 = zero8(), O1 = zero8(), O2 = zero8(), O3 = zero8();
  float mrow[8], lrow[8];
#pragma unroll
  for (int r = 0; r < 8; ++r) { mrow[r] = -1e30f; lrow[r] = 0.0f; }

  __shared__ __bf16 Plds[16][32];

  // K tile buffer A: chunk 0
  v16bf ka0 = loadB16(Kp + (size_t)(m)      * DKk, 0,  half);
  v16bf ka1 = loadB16(Kp + (size_t)(m)      * DKk, 32, half);
  v16bf ka2 = loadB16(Kp + (size_t)(16 + m) * DKk, 0,  half);
  v16bf ka3 = loadB16(Kp + (size_t)(16 + m) * DKk, 32, half);
  v16bf kb0, kb1, kb2, kb3;  // buffer B, filled by prefetch

  // Unroll-by-2 over key chunks (Tt/32 = 64 chunks, even)
  for (int c0 = 0; c0 < Tt; c0 += 64) {
    attn_chunk(c0, m, half, qa0, qa1, ka0, ka1, ka2, ka3,
               c0 + 32, kb0, kb1, kb2, kb3,
               Kp, Vp, mp, Plds, O0, O1, O2, O3, mrow, lrow);
    attn_chunk(c0 + 32, m, half, qa0, qa1, kb0, kb1, kb2, kb3,
               c0 + 64, ka0, ka1, ka2, ka3,
               Kp, Vp, mp, Plds, O0, O1, O2, O3, mrow, lrow);
  }

  // Normalize and store ctx[b, q, h*64+d] (bf16)
#pragma unroll
  for (int r = 0; r < 8; ++r) lrow[r] = 1.0f / lrow[r];
#pragma unroll
  for (int r = 0; r < 8; ++r) {
    const int row = q0 + r + 8 * half;
    const size_t base = ((size_t)b * Tt + row) * Cc + h * DKk + m;
    ctx[base +  0] = f2bf(O0[r] * lrow[r]);
    ctx[base + 16] = f2bf(O1[r] * lrow[r]);
    ctx[base + 32] = f2bf(O2[r] * lrow[r]);
    ctx[base + 48] = f2bf(O3[r] * lrow[r]);
  }
}

// ---------------------------------------------------------------------------
extern "C" void kernel_launch(void* const* d_in, const int* in_sizes, int n_in,
                              void* d_out, int out_size, void* d_ws, size_t ws_size,
                              hipStream_t stream) {
  (void)in_sizes; (void)n_in; (void)out_size;
  const float* query = (const float*)d_in[0];
  const float* keyt  = (const float*)d_in[1];
  const float* value = (const float*)d_in[2];
  const float* Wq = (const float*)d_in[3];  const float* bq = (const float*)d_in[4];
  const float* Wk = (const float*)d_in[5];  const float* bk = (const float*)d_in[6];
  const float* Wv = (const float*)d_in[7];  const float* bv = (const float*)d_in[8];
  const float* Wo = (const float*)d_in[9];  const float* bo = (const float*)d_in[10];
  const unsigned char* kpm = (const unsigned char*)d_in[11];
  float* out = (float*)d_out;

  const size_t xBytes = (size_t)Bb * Tt * Cc * sizeof(__bf16);   // 8 MB
  const size_t wBytes = (size_t)Cc * Cc * sizeof(__bf16);        // 2 MB
  const size_t need = 6 * xBytes + 4 * wBytes;                   // 56 MB
  if (ws_size < need) return;
  char* ws = (char*)d_ws;
  __bf16* Xq  = (__bf16*)(ws);                       // reused as ctx later
  __bf16* Xk  = (__bf16*)(ws + 1 * xBytes);
  __bf16* Xv  = (__bf16*)(ws + 2 * xBytes);
  __bf16* Wqb = (__bf16*)(ws + 3 * xBytes);
  __bf16* Wkb = (__bf16*)(ws + 3 * xBytes + 1 * wBytes);
  __bf16* Wvb = (__bf16*)(ws + 3 * xBytes + 2 * wBytes);
  __bf16* Wob = (__bf16*)(ws + 3 * xBytes + 3 * wBytes);
  __bf16* Qw  = (__bf16*)(ws + 3 * xBytes + 4 * wBytes);
  __bf16* Kw  = (__bf16*)(ws + 4 * xBytes + 4 * wBytes);
  __bf16* Vt  = (__bf16*)(ws + 5 * xBytes + 4 * wBytes);
  __bf16* ctx = Xq;  // Xq is dead after the Q projection

  const int M = Bb * Tt;  // 4096
  {
    dim3 blk(256);
    dim3 gx((M * (Cc / 4) + 255) / 256);
    dim3 gw((Cc * (Cc / 4) + 255) / 256);
    cvt_rows<<<gx, blk, 0, stream>>>(query, Xq, M, 1);
    cvt_rows<<<gx, blk, 0, stream>>>(keyt,  Xk, M, 1);
    cvt_rows<<<gx, blk, 0, stream>>>(value, Xv, M, 1);
    cvt_rows<<<gw, blk, 0, stream>>>(Wq, Wqb, Cc, 0);
    cvt_rows<<<gw, blk, 0, stream>>>(Wk, Wkb, Cc, 0);
    cvt_rows<<<gw, blk, 0, stream>>>(Wv, Wvb, Cc, 0);
    cvt_rows<<<gw, blk, 0, stream>>>(Wo, Wob, Cc, 0);
  }

  dim3 gblk(128), ggrid(M / 128, Cc / 64);
  const float qscale = 0.125f;  // 1/sqrt(DK)
  gemm_bf16<<<ggrid, gblk, 0, stream>>>(Xq, Wqb, bq, qscale, 0, Qw, nullptr);
  gemm_bf16<<<ggrid, gblk, 0, stream>>>(Xk, Wkb, bk, 1.0f,   0, Kw, nullptr);
  gemm_bf16<<<ggrid, gblk, 0, stream>>>(Xv, Wvb, bv, 1.0f,   1, Vt, nullptr);

  dim3 fgrid(Tt / 16, Bb * Hh), fblk(32);
  flash_attn<<<fgrid, fblk, 0, stream>>>(Qw, Kw, Vt, kpm, ctx);

  gemm_bf16<<<ggrid, gblk, 0, stream>>>(ctx, Wob, bo, 1.0f, 2, nullptr, out);
}
